// Morphology_35716948034282
// MI455X (gfx1250) — compile-verified, matrix operand order checked by reading, not run on hardware
//
#include <hip/hip_runtime.h>
#include <hip/hip_bf16.h>

typedef __attribute__((ext_vector_type(16))) __bf16       v16bf;
typedef __attribute__((ext_vector_type(8)))  float        v8f;
typedef __attribute__((ext_vector_type(4)))  unsigned int v4u;
typedef __attribute__((ext_vector_type(8)))  unsigned int v8u;

#define KW     5
#define BETA   15.0f
#define SHIFT  20.0f
#define HH     128
#define WW     128

// Block covers 32 rows x 64 cols of output; 8 waves, each one 16x16 tile.
#define TROWS   36   // 32 + 4 halo rows of E
#define TCOLS   68   // 64 + 4 halo cols of E
#define LSTRIDE 56   // column height (col-major LDS): >=48 so K=0..31 reads are
                     // in-bounds, multiple of 8 for 16B-aligned columns,
                     // 112B stride -> conflict-free 16-lane column access
#define PADROWS (LSTRIDE - TROWS)

__global__ __launch_bounds__(256)
void morph_softdilate_kernel(const float* __restrict__ x,
                             const float* __restrict__ wgt,
                             float* __restrict__ out) {
    __shared__ __attribute__((aligned(32))) __bf16 Elds[TCOLS * LSTRIDE];
    __shared__ __attribute__((aligned(32))) __bf16 Atab[KW * 32 * 16]; // [dj][lane][e]
    __shared__ float coef[32];

    const int tid = threadIdx.x;
    const int bx  = blockIdx.x;           // W/64 tiles
    const int by  = blockIdx.y;           // H/32 tiles
    const int bc  = blockIdx.z;           // b*C + c
    const int ch  = bc & 63;              // C = 64
    const int gy0 = by * 32;
    const int gx0 = bx * 64;

    const float* xin = x + (size_t)bc * (HH * WW);

    // Per-channel tap coefficients a[di*5+dj] = exp(beta * w)
    if (tid < 25) coef[tid] = __expf(wgt[ch * 25 + tid] * BETA);
    __syncthreads();

    // --- Precompute A matrices (banded Toeplitz) in WMMA A-register layout.
    // A layout (ISA 7.12.2, 16-bit A 16x32): lane<16 -> M=lane, elem e -> K = e<8 ? e : e+8
    //                                        lane>=16 -> M=lane-16, K += 8
    for (int i = tid; i < KW * 32 * 16; i += 256) {
        int dj = i >> 9;                 // /512
        int r  = i & 511;
        int ln = r >> 4;
        int e  = r & 15;
        int mm = ln & 15;
        int k  = ((e < 8) ? e : e + 8) + ((ln >= 16) ? 8 : 0);
        int di = k - mm;
        float av = (k < 20 && di >= 0 && di < KW) ? coef[di * KW + dj] : 0.0f;
        Atab[i] = (__bf16)av;
    }

    // --- Load halo tile, fuse exp: E = exp(beta*x - SHIFT); zero-padded x -> e^{-SHIFT}.
    // Column-major: Elds[col*LSTRIDE + row].
    for (int i = tid; i < TROWS * TCOLS; i += 256) {
        int r  = i / TCOLS;
        int cl = i - r * TCOLS;
        int gr = gy0 + r  - 2;
        int gc = gx0 + cl - 2;
        float v = 0.0f;
        if ((unsigned)gr < HH && (unsigned)gc < WW) v = xin[gr * WW + gc];
        Elds[cl * LSTRIDE + r] = (__bf16)__expf(v * BETA - SHIFT);
    }
    // Zero the pad rows (K>=20 region reads them; A is zero there, but values
    // must be finite: NaN * 0 -> NaN under WMMA rules).
    for (int i = tid; i < TCOLS * PADROWS; i += 256) {
        int cl = i / PADROWS;
        int r  = TROWS + (i - cl * PADROWS);
        Elds[cl * LSTRIDE + r] = (__bf16)0.0f;
    }
    __syncthreads();

    const int wave  = tid >> 5;
    const int lane  = tid & 31;
    const int ty    = (wave >> 2) * 16;   // 0 or 16
    const int tx    = (wave & 3)  * 16;   // 0..48
    const int m     = lane & 15;
    const int hioff = (lane >= 16) ? 8 : 0;

    v8f acc = {};  // f32 C/D accumulator

    #pragma unroll
    for (int dj = 0; dj < KW; ++dj) {
        // A: one aligned 32B LDS read per lane (pre-laid-out)
        const v8u* ap = (const v8u*)&Atab[(dj * 32 + lane) << 4];
        v16bf a = __builtin_bit_cast(v16bf, *ap);

        // B: E patch column tx+m+dj, rows ty+hioff + {0..7, 16..23} -> 2x b128
        const v4u* bp = (const v4u*)&Elds[(tx + m + dj) * LSTRIDE + ty + hioff];
        v4u blo = bp[0];
        v4u bhi = bp[2];
        v8u braw = {blo.x, blo.y, blo.z, blo.w, bhi.x, bhi.y, bhi.z, bhi.w};
        v16bf b = __builtin_bit_cast(v16bf, braw);

        // D = A x B + C  (f32 accumulate across the 5 kernel columns)
        acc = __builtin_amdgcn_wmma_f32_16x16x32_bf16(
                  false, a, false, b, (short)0, acc, false, false);
    }

    // C/D layout: VGPR r -> M = r (+8 for hi lanes), N = lane&15
    float* op = out + (size_t)bc * (HH * WW);
    const int row0 = gy0 + ty + hioff;
    const int colg = gx0 + tx + m;
    #pragma unroll
    for (int r = 0; r < 8; ++r) {
        float s = acc[r];
        float y = (__logf(s) + SHIFT) * (1.0f / BETA);
        op[(row0 + r) * WW + colg] = y;
    }
}

extern "C" void kernel_launch(void* const* d_in, const int* in_sizes, int n_in,
                              void* d_out, int out_size, void* d_ws, size_t ws_size,
                              hipStream_t stream) {
    const float* x   = (const float*)d_in[0];   // (4, 64, 128, 128) f32
    const float* wgt = (const float*)d_in[1];   // (64, 5, 5) f32
    float* out = (float*)d_out;

    const int bc = in_sizes[0] / (HH * WW);     // B*C = 256
    dim3 grid(WW / 64, HH / 32, bc);            // (2, 4, 256)
    dim3 block(256);
    morph_softdilate_kernel<<<grid, block, 0, stream>>>(x, wgt, out);
}
